// EncoderBlock_39779987096249
// MI455X (gfx1250) — matrix operand order repaired and checked by reading, not compiled, not run on hardware
//
#include <hip/hip_runtime.h>
#include <hip/hip_bf16.h>

// ---------------------------------------------------------------------------
// Types for CDNA5 WMMA (wave32): v_wmma_f32_16x16x32_f16
// ---------------------------------------------------------------------------
typedef _Float16 h8   __attribute__((ext_vector_type(8)));
typedef _Float16 v16h __attribute__((ext_vector_type(16)));
typedef float    v8f  __attribute__((ext_vector_type(8)));
typedef unsigned int u32x4 __attribute__((ext_vector_type(4)));
typedef int          i32x4 __attribute__((ext_vector_type(4)));
typedef int          i32x8 __attribute__((ext_vector_type(8)));

#if __has_builtin(__builtin_amdgcn_tensor_load_to_lds)
#define HAVE_TDM 1
#else
#define HAVE_TDM 0
#endif

static __device__ __forceinline__ v16h join8(h8 lo, h8 hi) {
  return __builtin_shufflevector(lo, hi, 0,1,2,3,4,5,6,7,8,9,10,11,12,13,14,15);
}

static __device__ __forceinline__ v8f wmma_f16(v16h a, v16h b, v8f c) {
  // (neg_a, A, neg_b, B, c_mod, C, reuse_a, reuse_b)
  return __builtin_amdgcn_wmma_f32_16x16x32_f16(false, a, false, b, (short)0, c,
                                                false, false);
}

#if HAVE_TDM
// ---------------------------------------------------------------------------
// TDM: 2D f16 tile (tile_d0 x tile_d1 elements) Global -> LDS with LDS row
// padding (pad codes per CDNA5 ISA):
//   pad_interval code c -> 2^(c+1) DWORDs before pad; pad_amount code c ->
//   (c+1) DWORDs of pad.
// ---------------------------------------------------------------------------
static __device__ __forceinline__ void tdm_load_tile_f16(
    unsigned lds_byte_off, const _Float16* gptr, unsigned tensor_d0,
    unsigned tensor_d1, unsigned tile_d0, unsigned tile_d1,
    unsigned stride_elems, unsigned pad_interval_code, unsigned pad_amount_code) {
  unsigned long long ga = (unsigned long long)(size_t)gptr;
  u32x4 g0;
  g0[0] = 1u;                                    // count=1 (valid user D#)
  g0[1] = lds_byte_off;                          // lds_addr
  g0[2] = (unsigned)ga;                          // global_addr[31:0]
  g0[3] = (unsigned)((ga >> 32) & 0x01FFFFFFu)   // global_addr[56:32]
          | 0x80000000u;                         // type=2 ("image")
  i32x8 g1;
  g1[0] = (int)((1u << 16)                       // data_size = 1 -> 2 bytes
              | (1u << 20)                       // pad_enable
              | (pad_interval_code << 22)
              | (pad_amount_code << 25));
  g1[1] = (int)((tensor_d0 & 0xFFFFu) << 16);                       // dim0 lo
  g1[2] = (int)(((tensor_d0 >> 16) & 0xFFFFu) |
                ((tensor_d1 & 0xFFFFu) << 16));                     // dim0 hi|dim1 lo
  g1[3] = (int)(((tensor_d1 >> 16) & 0xFFFFu) |
                ((tile_d0 & 0xFFFFu) << 16));                       // dim1 hi|tile0
  g1[4] = (int)(tile_d1 & 0xFFFFu);                                 // tile1, tile2=0
  g1[5] = (int)stride_elems;                                        // dim0_stride lo
  g1[6] = 0;
  g1[7] = 0;
  i32x4 z4 = {0, 0, 0, 0};
#if __clang_major__ >= 23
  i32x8 z8 = {0, 0, 0, 0, 0, 0, 0, 0};
  __builtin_amdgcn_tensor_load_to_lds(g0, g1, z4, z4, z8, 0);
#else
  __builtin_amdgcn_tensor_load_to_lds(g0, g1, z4, z4, 0);
#endif
}
#endif  // HAVE_TDM

// ---------------------------------------------------------------------------
// f32 -> f16 conversion
// ---------------------------------------------------------------------------
__global__ void cvt_kernel(const float* __restrict__ src, _Float16* __restrict__ dst,
                           int n) {
  int i = blockIdx.x * 256 + threadIdx.x;
  if (i < n) dst[i] = (_Float16)src[i];
}

// ---------------------------------------------------------------------------
// LayerNorm over D=768, f16 output. One block (256 threads) per row.
// ---------------------------------------------------------------------------
__global__ __launch_bounds__(256) void ln_kernel(const float* __restrict__ x,
                                                 const float* __restrict__ g,
                                                 const float* __restrict__ bta,
                                                 _Float16* __restrict__ out) {
  int row = blockIdx.x;
  const float* xr = x + (size_t)row * 768;
  float v[3], s1 = 0.f, s2 = 0.f;
#pragma unroll
  for (int j = 0; j < 3; ++j) {
    v[j] = xr[threadIdx.x + j * 256];
    s1 += v[j]; s2 += v[j] * v[j];
  }
#pragma unroll
  for (int m = 1; m <= 16; m <<= 1) {
    s1 += __shfl_xor(s1, m, 32);
    s2 += __shfl_xor(s2, m, 32);
  }
  __shared__ float w1[8], w2[8];
  int w = threadIdx.x >> 5, lane = threadIdx.x & 31;
  if (lane == 0) { w1[w] = s1; w2[w] = s2; }
  __syncthreads();
  s1 = 0.f; s2 = 0.f;
#pragma unroll
  for (int i = 0; i < 8; ++i) { s1 += w1[i]; s2 += w2[i]; }
  float mean = s1 * (1.0f / 768.0f);
  float var  = s2 * (1.0f / 768.0f) - mean * mean;
  float inv  = rsqrtf(var + 1e-5f);
  _Float16* orow = out + (size_t)row * 768;
#pragma unroll
  for (int j = 0; j < 3; ++j) {
    int i = threadIdx.x + j * 256;
    orow[i] = (_Float16)((v[j] - mean) * inv * g[i] + bta[i]);
  }
}

// ---------------------------------------------------------------------------
// GEMM: out[M,N] = A16[M,K] @ W16[N,K]^T + bias (+gelu) (+f32 residual)
// Block 256 threads = 8 waves; block tile 128x128; wave tile 32x64.
// K advances in 64-wide slabs staged in LDS by the Tensor Data Mover with
// double buffering: wave 0 issues the next slab's two tensor_load_to_lds
// right after the stage barrier, WMMAs consume the current slab, and the
// only synchronization per stage is (wave0: s_wait_tensorcnt 0) + barrier.
// TDM pads LDS rows 128B->144B so 16-row-strided ds_load_b128 is
// bank-conflict-free (stride 36 dwords).
// ---------------------------------------------------------------------------
template <bool GELU, bool F16OUT, bool RES>
__global__ __launch_bounds__(256) void gemm_kernel(const _Float16* __restrict__ A,
                                                   const _Float16* __restrict__ W,
                                                   const float* __restrict__ bias,
                                                   const float* __restrict__ res,
                                                   float* __restrict__ outF,
                                                   _Float16* __restrict__ outH,
                                                   int M, int N, int K) {
  constexpr int LDT = 72;  // 64 halves + 8 halves TDM pad
  __shared__ _Float16 Atile[2][128 * LDT];
  __shared__ _Float16 Btile[2][128 * LDT];

  int lane = threadIdx.x & 31;
  int w = threadIdx.x >> 5;
  int wm = w >> 1;        // 0..3 : 32-row slice
  int wn = w & 1;         // 0..1 : 64-col slice
  int m0 = blockIdx.x * 128;
  int n0 = blockIdx.y * 128;
  int k0a = (lane & 16) ? 16 : 0;  // A-frag K offset
  int kbb = (lane & 16) ? 16 : 0;  // B-frag K offset
  (void)M;

  v8f acc[2][4] = {};
  const int nstage = K >> 6;  // K / 64
  int cur = 0;

#if HAVE_TDM
  if (w == 0) {
    tdm_load_tile_f16((unsigned)(size_t)(void*)Atile[0], A + (size_t)m0 * K,
                      (unsigned)K, (unsigned)M, 64u, 128u, (unsigned)K, 4u, 3u);
    tdm_load_tile_f16((unsigned)(size_t)(void*)Btile[0], W + (size_t)n0 * K,
                      (unsigned)K, (unsigned)N, 64u, 128u, (unsigned)K, 4u, 3u);
  }
#endif

  for (int s = 0; s < nstage; ++s) {
#if HAVE_TDM
    if (w == 0) __builtin_amdgcn_s_wait_tensorcnt(0);
    __syncthreads();  // slab s visible; stage s-1 reads done everywhere
    if (w == 0 && s + 1 < nstage) {
      int kb = (s + 1) << 6;
      tdm_load_tile_f16((unsigned)(size_t)(void*)Atile[cur ^ 1],
                        A + (size_t)m0 * K + kb, (unsigned)K, (unsigned)M,
                        64u, 128u, (unsigned)K, 4u, 3u);
      tdm_load_tile_f16((unsigned)(size_t)(void*)Btile[cur ^ 1],
                        W + (size_t)n0 * K + kb, (unsigned)K, (unsigned)N,
                        64u, 128u, (unsigned)K, 4u, 3u);
    }
#else
    __syncthreads();  // previous stage's reads done
    {
      int kb = s << 6;
      int rowc = threadIdx.x & 127;
      int seg = (threadIdx.x >> 7) * 32;  // 0 or 32 halves
      const _Float16* sa = A + (size_t)(m0 + rowc) * K + kb + seg;
      _Float16* da = Atile[0] + rowc * LDT + seg;
#pragma unroll
      for (int q = 0; q < 4; ++q) *(h8*)(da + q * 8) = *(const h8*)(sa + q * 8);
      const _Float16* sb = W + (size_t)(n0 + rowc) * K + kb + seg;
      _Float16* db = Btile[0] + rowc * LDT + seg;
#pragma unroll
      for (int q = 0; q < 4; ++q) *(h8*)(db + q * 8) = *(const h8*)(sb + q * 8);
    }
    __syncthreads();
#endif

    const _Float16* At = Atile[cur];
    const _Float16* Bt = Btile[cur];
#pragma unroll
    for (int kk = 0; kk < 64; kk += 32) {
      v16h af[2], bf[4];
#pragma unroll
      for (int mt = 0; mt < 2; ++mt) {
        const _Float16* p = At + (wm * 32 + mt * 16 + (lane & 15)) * LDT + kk + k0a;
        af[mt] = join8(*(const h8*)p, *(const h8*)(p + 8));
      }
#pragma unroll
      for (int nt = 0; nt < 4; ++nt) {
        const _Float16* p = Bt + (wn * 64 + nt * 16 + (lane & 15)) * LDT + kk + kbb;
        bf[nt] = join8(*(const h8*)p, *(const h8*)(p + 8));
      }
#pragma unroll
      for (int mt = 0; mt < 2; ++mt)
#pragma unroll
        for (int nt = 0; nt < 4; ++nt)
          acc[mt][nt] = wmma_f16(af[mt], bf[nt], acc[mt][nt]);
    }
#if HAVE_TDM
    cur ^= 1;
#endif
  }

  int rbase = (lane & 16) ? 8 : 0;  // C/D layout: VGPR r -> row r (+8 hi half)
#pragma unroll
  for (int mt = 0; mt < 2; ++mt) {
#pragma unroll
    for (int nt = 0; nt < 4; ++nt) {
      int col = n0 + wn * 64 + nt * 16 + (lane & 15);
      float bc = bias[col];
#pragma unroll
      for (int r = 0; r < 8; ++r) {
        int row = m0 + wm * 32 + mt * 16 + rbase + r;
        float vv = acc[mt][nt][r] + bc;
        if (GELU) vv = 0.5f * vv * (1.0f + erff(vv * 0.70710678118654752f));
        if (RES)  vv += res[(size_t)row * N + col];
        if (F16OUT) outH[(size_t)row * N + col] = (_Float16)vv;
        else        outF[(size_t)row * N + col] = vv;
      }
    }
  }
}

// ---------------------------------------------------------------------------
// Flash attention over qkv (f16, row layout [3][12][64] per token).
// Block = 128 threads (4 waves); one block per (batch*head, 64-query tile);
// wave owns a 16-query tile. Streams 32-key chunks with online softmax.
// ---------------------------------------------------------------------------
__global__ __launch_bounds__(128) void attn_kernel(const _Float16* __restrict__ qkv,
                                                   _Float16* __restrict__ ctx) {
  const int N = 1024, ROW = 2304;
  int bh = blockIdx.x;
  int b = bh / 12, h = bh % 12;
  int w = threadIdx.x >> 5;
  int lane = threadIdx.x & 31;
  int qbase = blockIdx.y * 64 + w * 16;
  const _Float16* base = qkv + (size_t)b * N * ROW;

  int k0a  = (lane & 16) ? 16 : 0;   // A-frag K offset
  int kbb  = (lane & 16) ? 16 : 0;   // B-frag K offset
  int rbase = (lane & 16) ? 8 : 0;   // C/D row offset

  __shared__ _Float16 VT[64 * 32];       // V^T chunk: [d][key], shared by block
  __shared__ _Float16 PB[4][16 * 32];    // per-wave P tile (queries x 32 keys)

  // Q fragments (K = head_dim 64 -> 2 fragments)
  v16h qf[2];
  {
    const _Float16* qp = base + (size_t)(qbase + (lane & 15)) * ROW + h * 64;
#pragma unroll
    for (int kk = 0; kk < 2; ++kk) {
      const _Float16* p = qp + kk * 32 + k0a;
      qf[kk] = join8(*(const h8*)p, *(const h8*)(p + 8));
    }
  }

  float mrow[8], lrow[8];
  v8f accO[4] = {};
#pragma unroll
  for (int r = 0; r < 8; ++r) { mrow[r] = -1e30f; lrow[r] = 0.f; }
  const float scale = 0.125f;  // 64^-0.5

  int tok = threadIdx.x & 31;  // staging: token within chunk
  int dq  = threadIdx.x >> 5;  // staging: 16-dim group

  for (int c = 0; c < 32; ++c) {
    int keybase = c * 32;
    __syncthreads();  // previous chunk's LDS reads complete before overwrite

    // Prefetch next chunk's K and V rows (global_prefetch_b8)
    if (c < 31) {
      const _Float16* nk =
          base + (size_t)(keybase + 32 + tok) * ROW + 768 + h * 64;
      __builtin_prefetch(nk, 0, 1);
      __builtin_prefetch(nk + 768, 0, 1);  // V row = K row + 768 halves
    }

    // Cooperative V^T staging: 32 tokens x 64 dims
    {
      const _Float16* vp =
          base + (size_t)(keybase + tok) * ROW + 1536 + h * 64 + dq * 16;
      h8 lo = *(const h8*)vp;
      h8 hi = *(const h8*)(vp + 8);
#pragma unroll
      for (int j = 0; j < 8; ++j) VT[(dq * 16 + j) * 32 + tok] = lo[j];
#pragma unroll
      for (int j = 0; j < 8; ++j) VT[(dq * 16 + 8 + j) * 32 + tok] = hi[j];
    }

    // S = (Q @ K^T) * scale for 16 queries x 32 keys (two 16x16 tiles)
    v8f s[2];
#pragma unroll
    for (int nt = 0; nt < 2; ++nt) {
      const _Float16* kp =
          base + (size_t)(keybase + nt * 16 + (lane & 15)) * ROW + 768 + h * 64;
      v8f a = {};
#pragma unroll
      for (int kk = 0; kk < 2; ++kk) {
        const _Float16* p = kp + kk * 32 + kbb;
        v16h kf = join8(*(const h8*)p, *(const h8*)(p + 8));
        a = wmma_f16(qf[kk], kf, a);
      }
      s[nt] = a;
    }

    // Online softmax: per-row stats are uniform within each 16-lane half.
    float p0[8], p1[8];
#pragma unroll
    for (int r = 0; r < 8; ++r) {
      float s0 = s[0][r] * scale, s1 = s[1][r] * scale;
      float mx = fmaxf(s0, s1);
#pragma unroll
      for (int m = 1; m <= 8; m <<= 1) mx = fmaxf(mx, __shfl_xor(mx, m, 32));
      float mnew  = fmaxf(mrow[r], mx);
      float alpha = __expf(mrow[r] - mnew);
      float e0 = __expf(s0 - mnew), e1 = __expf(s1 - mnew);
      float sum = e0 + e1;
#pragma unroll
      for (int m = 1; m <= 8; m <<= 1) sum += __shfl_xor(sum, m, 32);
      lrow[r] = lrow[r] * alpha + sum;
      mrow[r] = mnew;
#pragma unroll
      for (int t = 0; t < 4; ++t) accO[t][r] *= alpha;
      p0[r] = e0; p1[r] = e1;
    }

    // P (C-layout) -> LDS -> reload in A-layout
    _Float16* pb = PB[w];
#pragma unroll
    for (int r = 0; r < 8; ++r) {
      pb[(rbase + r) * 32 + (lane & 15)]      = (_Float16)p0[r];
      pb[(rbase + r) * 32 + 16 + (lane & 15)] = (_Float16)p1[r];
    }
    __syncthreads();  // P and VT visible

    const _Float16* pr = pb + (size_t)(lane & 15) * 32 + k0a;
    v16h pf = join8(*(const h8*)pr, *(const h8*)(pr + 8));

    // O += P @ V  (K = 32 keys, four 16-dim output tiles)
#pragma unroll
    for (int t = 0; t < 4; ++t) {
      const _Float16* vtp = VT + (size_t)(t * 16 + (lane & 15)) * 32 + kbb;
      v16h vf = join8(*(const h8*)vtp, *(const h8*)(vtp + 8));
      accO[t] = wmma_f16(pf, vf, accO[t]);
    }
  }

  // Normalize and write context (f16)
#pragma unroll
  for (int t = 0; t < 4; ++t) {
#pragma unroll
    for (int r = 0; r < 8; ++r) {
      int row = qbase + rbase + r;
      int col = h * 64 + t * 16 + (lane & 15);
      ctx[((size_t)b * N + row) * 768 + col] = (_Float16)(accO[t][r] / lrow[r]);
    }
  }
}

// ---------------------------------------------------------------------------
// Launch
// ---------------------------------------------------------------------------
extern "C" void kernel_launch(void* const* d_in, const int* in_sizes, int n_in,
                              void* d_out, int out_size, void* d_ws, size_t ws_size,
                              hipStream_t stream) {
  (void)in_sizes; (void)n_in; (void)out_size; (void)ws_size;
  const float* x      = (const float*)d_in[0];
  const float* ln1_g  = (const float*)d_in[1];
  const float* ln1_b  = (const float*)d_in[2];
  const float* qkv_w  = (const float*)d_in[3];
  const float* qkv_b  = (const float*)d_in[4];
  const float* proj_w = (const float*)d_in[5];
  const float* proj_b = (const float*)d_in[6];
  const float* ln2_g  = (const float*)d_in[7];
  const float* ln2_b  = (const float*)d_in[8];
  const float* fc1_w  = (const float*)d_in[9];
  const float* fc1_b  = (const float*)d_in[10];
  const float* fc2_w  = (const float*)d_in[11];
  const float* fc2_b  = (const float*)d_in[12];
  float* out = (float*)d_out;

  const int M = 8192;          // 8 * 1024 tokens
  char* ws = (char*)d_ws;
  _Float16* w16_qkv  = (_Float16*)(ws);                         // 2304*768
  _Float16* w16_proj = (_Float16*)(ws + 3538944);               //  768*768
  _Float16* w16_fc1  = (_Float16*)(ws + 3538944 + 1179648);     // 3072*768
  _Float16* w16_fc2  = (_Float16*)(ws + 9437184);               //  768*3072
  _Float16* R1       = (_Float16*)(ws + 14155776);              // 8192*3072 (qkv16 / gelu16)
  _Float16* R2       = (_Float16*)(ws + 64487424);              // 8192*768  (ln out)
  _Float16* R3       = (_Float16*)(ws + 77070336);              // 8192*768  (ctx)

  // 1) weights f32 -> f16
  {
    int n;
    n = 2304 * 768;  cvt_kernel<<<(n + 255) / 256, 256, 0, stream>>>(qkv_w,  w16_qkv,  n);
    n =  768 * 768;  cvt_kernel<<<(n + 255) / 256, 256, 0, stream>>>(proj_w, w16_proj, n);
    n = 3072 * 768;  cvt_kernel<<<(n + 255) / 256, 256, 0, stream>>>(fc1_w,  w16_fc1,  n);
    n =  768 * 3072; cvt_kernel<<<(n + 255) / 256, 256, 0, stream>>>(fc2_w,  w16_fc2,  n);
  }

  // 2) LN1
  ln_kernel<<<M, 256, 0, stream>>>(x, ln1_g, ln1_b, R2);

  // 3) QKV: [8192,2304] = R2 @ qkv_w^T + b  (f16 out)
  gemm_kernel<false, true, false><<<dim3(M / 128, 2304 / 128), 256, 0, stream>>>(
      R2, w16_qkv, qkv_b, nullptr, nullptr, R1, M, 2304, 768);

  // 4) attention -> ctx (f16)
  attn_kernel<<<dim3(8 * 12, 1024 / 64), 128, 0, stream>>>(R1, R3);

  // 5) x2 = x + ctx @ proj_w^T + b  (f32, into d_out)
  gemm_kernel<false, false, true><<<dim3(M / 128, 768 / 128), 256, 0, stream>>>(
      R3, w16_proj, proj_b, x, out, nullptr, M, 768, 768);

  // 6) LN2
  ln_kernel<<<M, 256, 0, stream>>>(out, ln2_g, ln2_b, R2);

  // 7) gelu(R2 @ fc1_w^T + b) (f16 out)
  gemm_kernel<true, true, false><<<dim3(M / 128, 3072 / 128), 256, 0, stream>>>(
      R2, w16_fc1, fc1_b, nullptr, nullptr, R1, M, 3072, 768);

  // 8) d_out = x2 + R1 @ fc2_w^T + b  (in-place residual, element-disjoint)
  gemm_kernel<false, false, true><<<dim3(M / 128, 768 / 128), 256, 0, stream>>>(
      R1, w16_fc2, fc2_b, out, out, nullptr, M, 768, 3072);
}
